// CausalSelfAttention_20890720928009
// MI455X (gfx1250) — matrix operand order repaired and checked by reading, not compiled, not run on hardware
//
#include <hip/hip_runtime.h>
#include <hip/hip_bf16.h>

typedef __bf16 bf16;
typedef __attribute__((ext_vector_type(16))) __bf16 v16bf;
typedef __attribute__((ext_vector_type(8)))  __bf16 v8bf;
typedef __attribute__((ext_vector_type(8)))  float  v8f;

#define B_ 4
#define T_ 2048
#define C_ 1024
#define H_ 16
#define D_ 64

static __device__ __forceinline__ bf16 to_bf(float f) {
    unsigned u = __builtin_bit_cast(unsigned, f);
    u += 0x7fffu + ((u >> 16) & 1u);            // round-to-nearest-even
    unsigned short h = (unsigned short)(u >> 16);
    return __builtin_bit_cast(bf16, h);
}

union AFrag { v16bf v; v8bf h[2]; };

// A-fragment (16x32 bf16, row-major source, stride ld elements).
// lanes 0-15: M=lane, K = {k0+0..7, k0+16..23}; lanes 16-31: M=lane-16, K = {k0+8..15, k0+24..31}
static __device__ __forceinline__ v16bf load_a(const bf16* __restrict__ src, int ld,
                                               int m0, int k0, int lane) {
    int half = lane >> 4;
    const bf16* p = src + (size_t)(m0 + (lane & 15)) * (size_t)ld + k0 + half * 8;
    AFrag f;
    f.h[0] = *(const v8bf*)(p);
    f.h[1] = *(const v8bf*)(p + 16);
    return f.v;
}

// B-fragment (32x16 bf16) from Bt stored N x K row-major (i.e. B column-major).
// lanes 0-15: N=lane, K = {k0+0..7, k0+16..23}; lanes 16-31: N=lane-16, K = {k0+8..15, k0+24..31}
static __device__ __forceinline__ v16bf load_bt(const bf16* __restrict__ bt, int ld,
                                                int n0, int k0, int lane) {
    int half = lane >> 4;
    const bf16* p = bt + (size_t)(n0 + (lane & 15)) * (size_t)ld + k0 + half * 8;
    AFrag f;
    f.h[0] = *(const v8bf*)(p);
    f.h[1] = *(const v8bf*)(p + 16);
    return f.v;
}

static __device__ __forceinline__ v8f wmma_bf16(v16bf a, v16bf b, v8f c) {
    return __builtin_amdgcn_wmma_f32_16x16x32_bf16(false, a, false, b, (short)0, c,
                                                   false, false);
}

// ---- GEMM mainloop: one wave computes a 16(M) x 64(N) tile, K-loop step 32 ----
static __device__ __forceinline__ void gemm_tile_16x64(const bf16* __restrict__ A, int lda,
                                                       const bf16* __restrict__ Bt, int ldb,
                                                       int m0, int n0, int K, int lane,
                                                       v8f acc[4]) {
    #pragma unroll
    for (int j = 0; j < 4; ++j) acc[j] = (v8f){};
    for (int k = 0; k < K; k += 32) {
        v16bf a = load_a(A, lda, m0, k, lane);
        #pragma unroll
        for (int j = 0; j < 4; ++j) {
            v16bf b = load_bt(Bt, ldb, n0 + j * 16, k, lane);
            acc[j] = wmma_bf16(a, b, acc[j]);
        }
    }
}

// ---------------- conversion kernels ----------------
__global__ void f32_to_bf16_kernel(const float* __restrict__ in, bf16* __restrict__ out,
                                   size_t n) {
    size_t i = (size_t)blockIdx.x * blockDim.x + threadIdx.x;
    if (i < n) out[i] = to_bf(in[i]);
}

// in: K x N (row-major), out: N x K (row-major) bf16
__global__ void transpose_to_bf16_kernel(const float* __restrict__ in, bf16* __restrict__ out,
                                         int K, int N) {
    size_t i = (size_t)blockIdx.x * blockDim.x + threadIdx.x;
    size_t total = (size_t)K * N;
    if (i >= total) return;
    int n = (int)(i / K);
    int k = (int)(i % K);
    out[(size_t)n * K + k] = to_bf(in[(size_t)k * N + n]);
}

// ---------------- fused QKV GEMM + bias + RoPE + head packing ----------------
// X: (B*T, C) bf16 ; Wt: (3C, C) bf16 ; bias: (3C) f32
// Q,K out: (B,H,T,D) bf16 ; V out transposed: (B,H,D,T) bf16
__global__ __launch_bounds__(256) void qkv_rope_kernel(const bf16* __restrict__ X,
                                                       const bf16* __restrict__ Wt,
                                                       const float* __restrict__ bias,
                                                       bf16* __restrict__ Q,
                                                       bf16* __restrict__ K,
                                                       bf16* __restrict__ Vt) {
    int lane = threadIdx.x & 31;
    int wave = threadIdx.x >> 5;
    int m0 = blockIdx.x * 128 + wave * 16;   // row in (B*T)
    int n0 = blockIdx.y * 64;                // col in (3C), aligned to one head
    v8f acc[4];
    gemm_tile_16x64(X, C_, Wt, C_, m0, n0, C_, lane, acc);

    int half  = lane >> 4;
    int which = n0 / C_;            // 0=q, 1=k, 2=v
    int h     = (n0 % C_) / D_;

    #pragma unroll
    for (int j = 0; j < 4; ++j) {
        float bv = bias[n0 + j * 16 + (lane & 15)];
        #pragma unroll
        for (int r = 0; r < 8; ++r) acc[j][r] += bv;
    }

    if (which == 2) {
        // V -> transposed store Vt[(bh*D + d)*T + t]
        #pragma unroll
        for (int j = 0; j < 4; ++j) {
            int d = j * 16 + (lane & 15);
            #pragma unroll
            for (int r = 0; r < 8; ++r) {
                int row = m0 + r + half * 8;
                int b = row / T_, t = row % T_;
                Vt[((size_t)(b * H_ + h) * D_ + d) * T_ + t] = to_bf(acc[j][r]);
            }
        }
    } else {
        bf16* Out = (which == 0) ? Q : K;
        // RoPE in-register: pair (d, d+32) == acc tiles (j, j+2)
        #pragma unroll
        for (int j = 0; j < 2; ++j) {
            int dlo = j * 16 + (lane & 15);          // < 32
            float inv = __powf(10000.0f, -(2.0f * (float)dlo) / (float)D_);
            #pragma unroll
            for (int r = 0; r < 8; ++r) {
                int row = m0 + r + half * 8;
                int t = row % T_;
                float c, s;
                __sincosf((float)t * inv, &c, &s);
                float lo = acc[j][r], hi = acc[j + 2][r];
                acc[j][r]     = lo * c - hi * s;
                acc[j + 2][r] = hi * c + lo * s;
            }
        }
        #pragma unroll
        for (int j = 0; j < 4; ++j) {
            int d = j * 16 + (lane & 15);
            #pragma unroll
            for (int r = 0; r < 8; ++r) {
                int row = m0 + r + half * 8;
                int b = row / T_, t = row % T_;
                Out[((size_t)(b * H_ + h) * T_ + t) * D_ + d] = to_bf(acc[j][r]);
            }
        }
    }
}

// ---------------- flash attention (one wave per 32-query block) ----------------
// Two 16-row M-tiles per wave: K/V fragments are loaded once per 32-key block
// and reused by both M-tiles (halves K/V read traffic, doubles wmma:vmem).
// Q,K: (B,H,T,D) bf16 ; Vt: (B,H,D,T) bf16 ; Y out: (B,T,C) bf16
__global__ __launch_bounds__(32) void flash_kernel(const bf16* __restrict__ Q,
                                                   const bf16* __restrict__ K,
                                                   const bf16* __restrict__ Vt,
                                                   bf16* __restrict__ Y) {
    __shared__ alignas(16) bf16 Plds[16 * 32];
    int lane = threadIdx.x & 31;
    int half = lane >> 4;
    int tBase = blockIdx.x * 32;
    int bh = blockIdx.y;
    int b = bh / H_, h = bh % H_;

    const bf16* Qp = Q + (size_t)bh * T_ * D_;
    const bf16* Kp = K + (size_t)bh * T_ * D_;
    const bf16* Vp = Vt + (size_t)bh * D_ * T_;

    // Q fragments for both M-tiles (D = 64 -> two K-slabs each)
    v16bf qf[2][2];
    #pragma unroll
    for (int mt = 0; mt < 2; ++mt) {
        qf[mt][0] = load_a(Qp, D_, tBase + mt * 16, 0, lane);
        qf[mt][1] = load_a(Qp, D_, tBase + mt * 16, 32, lane);
    }

    v8f o[2][4];
    float m[2][8], l[2][8];
    #pragma unroll
    for (int mt = 0; mt < 2; ++mt) {
        #pragma unroll
        for (int j = 0; j < 4; ++j) o[mt][j] = (v8f){};
        #pragma unroll
        for (int r = 0; r < 8; ++r) { m[mt][r] = -3.0e38f; l[mt][r] = 0.0f; }
    }

    int nBlocks = tBase / 32 + 1;   // keys 0 .. tBase+31
    for (int sb = 0; sb < nBlocks; ++sb) {
        int sBase = sb * 32;

        // hoisted K/V fragments, shared by both M-tiles
        v16bf kf00 = load_bt(Kp, D_, sBase,      0,  lane);
        v16bf kf01 = load_bt(Kp, D_, sBase,      32, lane);
        v16bf kf10 = load_bt(Kp, D_, sBase + 16, 0,  lane);
        v16bf kf11 = load_bt(Kp, D_, sBase + 16, 32, lane);
        v16bf vf[4];
        #pragma unroll
        for (int j = 0; j < 4; ++j) vf[j] = load_bt(Vp, T_, j * 16, sBase, lane);

        int sKey0 = sBase + (lane & 15);
        int sKey1 = sKey0 + 16;

        #pragma unroll
        for (int mt = 0; mt < 2; ++mt) {
            int tb = tBase + mt * 16;
            // S = Q @ K^T (two 16x16 tiles over 32 keys, K-dim = D = 64)
            v8f s0 = (v8f){}, s1 = (v8f){};
            s0 = wmma_bf16(qf[mt][0], kf00, s0);
            s0 = wmma_bf16(qf[mt][1], kf01, s0);
            s1 = wmma_bf16(qf[mt][0], kf10, s1);
            s1 = wmma_bf16(qf[mt][1], kf11, s1);

            bool full = (sBase + 31) <= tb;
            if (!full) {
                #pragma unroll
                for (int r = 0; r < 8; ++r) {
                    int trow = tb + r + half * 8;
                    if (sKey0 > trow) s0[r] = -1.0e30f;
                    if (sKey1 > trow) s1[r] = -1.0e30f;
                }
            }

            // online softmax per row (row M=r+8*half lives on 16 lanes of this half)
            #pragma unroll
            for (int r = 0; r < 8; ++r) {
                float rowm = fmaxf(s0[r], s1[r]);
                #pragma unroll
                for (int off = 1; off < 16; off <<= 1)
                    rowm = fmaxf(rowm, __shfl_xor(rowm, off, 32));
                float mn = fmaxf(m[mt][r], rowm);
                float scale = __expf(m[mt][r] - mn);
                float p0 = __expf(s0[r] - mn);
                float p1 = __expf(s1[r] - mn);
                float rsum = p0 + p1;
                #pragma unroll
                for (int off = 1; off < 16; off <<= 1)
                    rsum += __shfl_xor(rsum, off, 32);
                l[mt][r] = l[mt][r] * scale + rsum;
                m[mt][r] = mn;
                #pragma unroll
                for (int j = 0; j < 4; ++j) o[mt][j][r] *= scale;
                int row = r + half * 8;
                Plds[row * 32 + (lane & 15)]      = to_bf(p0);
                Plds[row * 32 + 16 + (lane & 15)] = to_bf(p1);
            }
            __syncthreads();   // single-wave WG: lowers to s_nop; orders LDS

            // O += P @ V   (A-frag of P from LDS, hoisted V fragments)
            v16bf pf = load_a(Plds, 32, 0, 0, lane);
            #pragma unroll
            for (int j = 0; j < 4; ++j)
                o[mt][j] = wmma_bf16(pf, vf[j], o[mt][j]);
            __syncthreads();
        }
    }

    // normalize and store to (B,T,C) bf16 for the output projection
    #pragma unroll
    for (int mt = 0; mt < 2; ++mt) {
        #pragma unroll
        for (int j = 0; j < 4; ++j) {
            int d = j * 16 + (lane & 15);
            #pragma unroll
            for (int r = 0; r < 8; ++r) {
                int t = tBase + mt * 16 + r + half * 8;
                float val = o[mt][j][r] / l[mt][r];
                Y[((size_t)b * T_ + t) * C_ + h * D_ + d] = to_bf(val);
            }
        }
    }
}

// ---------------- output projection ----------------
__global__ __launch_bounds__(256) void out_proj_kernel(const bf16* __restrict__ Ybf,
                                                       const bf16* __restrict__ Wt,
                                                       const float* __restrict__ bias,
                                                       float* __restrict__ out) {
    int lane = threadIdx.x & 31;
    int wave = threadIdx.x >> 5;
    int m0 = blockIdx.x * 128 + wave * 16;
    int n0 = blockIdx.y * 64;
    v8f acc[4];
    gemm_tile_16x64(Ybf, C_, Wt, C_, m0, n0, C_, lane, acc);
    int half = lane >> 4;
    #pragma unroll
    for (int j = 0; j < 4; ++j) {
        int col = n0 + j * 16 + (lane & 15);
        float bv = bias[col];
        #pragma unroll
        for (int r = 0; r < 8; ++r) {
            int row = m0 + r + half * 8;
            out[(size_t)row * C_ + col] = acc[j][r] + bv;
        }
    }
}

extern "C" void kernel_launch(void* const* d_in, const int* in_sizes, int n_in,
                              void* d_out, int out_size, void* d_ws, size_t ws_size,
                              hipStream_t stream) {
    const float* x     = (const float*)d_in[0];   // (B,T,C)
    const float* W_qkv = (const float*)d_in[1];   // (C,3C)
    const float* b_qkv = (const float*)d_in[2];   // (3C)
    const float* W_out = (const float*)d_in[3];   // (C,C)
    const float* b_out = (const float*)d_in[4];   // (C)
    float* out = (float*)d_out;

    char* ws = (char*)d_ws;
    size_t offX     = 0;                                          // 8192*1024 bf16
    size_t offWqkvt = offX     + (size_t)B_ * T_ * C_ * 2;        // 3072*1024 bf16
    size_t offWoutt = offWqkvt + (size_t)(3 * C_) * C_ * 2;       // 1024*1024 bf16
    size_t offQ     = offWoutt + (size_t)C_ * C_ * 2;             // B*H*T*D bf16
    size_t offK     = offQ     + (size_t)B_ * H_ * T_ * D_ * 2;
    size_t offV     = offK     + (size_t)B_ * H_ * T_ * D_ * 2;
    size_t offY     = offV     + (size_t)B_ * H_ * T_ * D_ * 2;   // (B,T,C) bf16

    bf16* Xbf   = (bf16*)(ws + offX);
    bf16* Wqkvt = (bf16*)(ws + offWqkvt);
    bf16* Woutt = (bf16*)(ws + offWoutt);
    bf16* Qb    = (bf16*)(ws + offQ);
    bf16* Kb    = (bf16*)(ws + offK);
    bf16* Vt    = (bf16*)(ws + offV);
    bf16* Ybf   = (bf16*)(ws + offY);

    // 1. x -> bf16
    {
        size_t n = (size_t)B_ * T_ * C_;
        f32_to_bf16_kernel<<<(unsigned)((n + 255) / 256), 256, 0, stream>>>(x, Xbf, n);
    }
    // 2. W_qkv^T, W_out^T -> bf16 (N x K row-major)
    {
        size_t n = (size_t)C_ * 3 * C_;
        transpose_to_bf16_kernel<<<(unsigned)((n + 255) / 256), 256, 0, stream>>>(
            W_qkv, Wqkvt, C_, 3 * C_);
    }
    {
        size_t n = (size_t)C_ * C_;
        transpose_to_bf16_kernel<<<(unsigned)((n + 255) / 256), 256, 0, stream>>>(
            W_out, Woutt, C_, C_);
    }
    // 3. fused QKV GEMM + bias + RoPE + packing
    qkv_rope_kernel<<<dim3((B_ * T_) / 128, (3 * C_) / 64), 256, 0, stream>>>(
        Xbf, Wqkvt, b_qkv, Qb, Kb, Vt);
    // 4. flash attention (32 query rows per wave)
    flash_kernel<<<dim3(T_ / 32, B_ * H_), 32, 0, stream>>>(Qb, Kb, Vt, Ybf);
    // 5. output projection + bias -> fp32
    out_proj_kernel<<<dim3((B_ * T_) / 128, C_ / 64), 256, 0, stream>>>(
        Ybf, Woutt, b_out, out);
}
